// Seq2SeqLSTM_6451040878771
// MI455X (gfx1250) — matrix operand order repaired and checked by reading, not compiled, not run on hardware
//
#include <hip/hip_runtime.h>
#include <hip/hip_bf16.h>
#include <math.h>

// ---------------- problem constants ----------------
#define B_   64
#define S_   256
#define T_   32
#define E_   512
#define H_   512
#define G4H  2048          // 4*H
#define VT_  32000
#define BOS_ 2

typedef __bf16 v16bf __attribute__((ext_vector_type(16)));
typedef __bf16 v8bf  __attribute__((ext_vector_type(8)));
typedef float  v8f   __attribute__((ext_vector_type(8)));

// Load A fragment (16-bit A 16x32) for one K-chunk:
// lane's 16 elements = one contiguous 16-half run at k0+(hi?16:0) (matches B).
__device__ __forceinline__ v16bf load_a_frag(const __bf16* __restrict__ rowp,
                                             int k0, bool hi) {
    return *(const v16bf*)(rowp + k0 + (hi ? 16 : 0));
}

// Load B fragment (16-bit B 32x16, contiguous k order, matches A): one contiguous 16-half run
// at k0+(hi?16:0) along the weight row (W stored [N,K] row-major).
__device__ __forceinline__ v16bf load_b_frag(const __bf16* __restrict__ rowp,
                                             int k0, bool hi) {
    return *(const v16bf*)(rowp + k0 + (hi ? 16 : 0));
}

// ---------------- fp32 -> bf16 weight conversion ----------------
__global__ void f32_to_bf16_kernel(const float* __restrict__ in,
                                   __bf16* __restrict__ out, int n) {
    int i = blockIdx.x * blockDim.x + threadIdx.x;
    int stride = gridDim.x * blockDim.x;
    for (; i < n; i += stride) out[i] = (__bf16)in[i];
}

// ---------------- per-step embedding gather + bf16 convert ----------------
// x_bf16[B,E] = bf16(emb[token[m], :])
__global__ __launch_bounds__(256)
void gather_embed_kernel(const float* __restrict__ emb,  // [V,E]
                         const int* __restrict__ tok,    // token source
                         int tok_stride, int tok_off, int use_bos,
                         __bf16* __restrict__ x) {       // [B,E]
    int i = blockIdx.x * blockDim.x + threadIdx.x;       // over B_*E_
    if (i >= B_ * E_) return;
    int m = i >> 9;            // E_ = 512
    int k = i & (E_ - 1);
    int t = use_bos ? BOS_ : tok[m * tok_stride + tok_off];
    x[i] = (__bf16)emb[(size_t)t * E_ + k];
}

// ---------------- fused LSTM layer-step ----------------
// grid = (H/16, B/16), block = 128 (4 waves; wave w computes gate w's 16x16 tile)
// gates = x @ Wih^T + h @ Whh^T + bih + bhh ; then i,f,g,o -> c,h update.
__global__ __launch_bounds__(128)
void lstm_step_kernel(const __bf16* __restrict__ xin,  // [B,E] bf16
                      const __bf16* __restrict__ Wih,  // [4H,E] bf16
                      const __bf16* __restrict__ Whh,  // [4H,H] bf16
                      const float* __restrict__ bih,   // [4H]
                      const float* __restrict__ bhh,   // [4H]
                      const __bf16* __restrict__ hprev,// [B,H] bf16 (read-only this step)
                      const float* __restrict__ cprev, // [B,H] (may alias cout)
                      __bf16* __restrict__ hout,       // [B,H] bf16 (other buffer)
                      float* __restrict__ cout)        // [B,H]
{
    __shared__ float sg[4][16][16];

    const int lane = threadIdx.x & 31;
    const int wave = threadIdx.x >> 5;           // gate id 0..3 (i,f,g,o)
    const int j0   = blockIdx.x * 16;            // hidden-col tile
    const int m0   = blockIdx.y * 16;            // batch-row tile
    const bool hi  = lane >= 16;
    const int  l15 = lane & 15;

    const int rowA = m0 + l15;                   // A-matrix row this lane holds
    const int colB = wave * H_ + j0 + l15;       // output col == W row this lane holds

    const __bf16* xrow  = xin   + (size_t)rowA * E_;
    const __bf16* hrow  = hprev + (size_t)rowA * H_;
    const __bf16* wihrw = Wih   + (size_t)colB * E_;
    const __bf16* whhrw = Whh   + (size_t)colB * H_;

    v8f acc = {};

    // ---- x @ Wih^T : K over E ----
    #pragma unroll 4
    for (int k0 = 0; k0 < E_; k0 += 32) {
        v16bf a = load_a_frag(xrow, k0, hi);
        v16bf b = load_b_frag(wihrw, k0, hi);
        acc = __builtin_amdgcn_wmma_f32_16x16x32_bf16(false, a, false, b,
                                                      (short)0, acc, false, false);
    }
    // ---- h @ Whh^T : K over H ----
    #pragma unroll 4
    for (int k0 = 0; k0 < H_; k0 += 32) {
        v16bf a = load_a_frag(hrow, k0, hi);
        v16bf b = load_b_frag(whhrw, k0, hi);
        acc = __builtin_amdgcn_wmma_f32_16x16x32_bf16(false, a, false, b,
                                                      (short)0, acc, false, false);
    }

    // bias + park gate tile in LDS (C/D layout: vgpr r -> row r + 8*(lane>=16), col = lane&15)
    const float bias = bih[colB] + bhh[colB];
    #pragma unroll
    for (int r = 0; r < 8; ++r) {
        int m = r + (hi ? 8 : 0);
        sg[wave][m][l15] = acc[r] + bias;
    }
    __syncthreads();

    // cell update: 256 elements, 128 threads
    for (int idx = threadIdx.x; idx < 256; idx += 128) {
        int m = idx >> 4, j = idx & 15;
        size_t o = (size_t)(m0 + m) * H_ + (j0 + j);
        float ig = sg[0][m][j], fg = sg[1][m][j];
        float gg = sg[2][m][j], og = sg[3][m][j];
        float cp = cprev[o];
        float si = 1.f / (1.f + __expf(-ig));
        float sf = 1.f / (1.f + __expf(-fg));
        float so = 1.f / (1.f + __expf(-og));
        float c = sf * cp + si * tanhf(gg);
        float h = so * tanhf(c);
        cout[o] = c;
        hout[o] = (__bf16)h;
    }
}

// ---------------- final FC: logits = h_top @ fcW^T + fcb ----------------
// grid = (VT/16, B/16), block = 32 (one wave per 16x16 tile)
__global__ __launch_bounds__(32)
void fc_kernel(const __bf16* __restrict__ htop,  // [B,H] bf16
               const __bf16* __restrict__ W,     // [VT,H] bf16
               const float* __restrict__ bias,   // [VT]
               float* __restrict__ out)          // [B,VT]
{
    const int lane = threadIdx.x & 31;
    const bool hi  = lane >= 16;
    const int  l15 = lane & 15;
    const int n0 = blockIdx.x * 16;
    const int m0 = blockIdx.y * 16;
    const int rowA = m0 + l15;
    const int colB = n0 + l15;

    const __bf16* hrow = htop + (size_t)rowA * H_;
    const __bf16* wrow = W    + (size_t)colB * H_;

    v8f acc = {};
    #pragma unroll 4
    for (int k0 = 0; k0 < H_; k0 += 32) {
        v16bf a = load_a_frag(hrow, k0, hi);
        v16bf b = load_b_frag(wrow, k0, hi);
        acc = __builtin_amdgcn_wmma_f32_16x16x32_bf16(false, a, false, b,
                                                      (short)0, acc, false, false);
    }
    float bs = bias[colB];
    #pragma unroll
    for (int r = 0; r < 8; ++r) {
        int m = r + (hi ? 8 : 0);
        out[(size_t)(m0 + m) * VT_ + colB] = acc[r] + bs;
    }
}

// ---------------- row-wise log-softmax in place ----------------
__global__ __launch_bounds__(256)
void log_softmax_kernel(float* __restrict__ out, int V) {
    __shared__ float red[256];
    float* p = out + (size_t)blockIdx.x * V;

    float mx = -INFINITY;
    for (int i = threadIdx.x; i < V; i += 256) mx = fmaxf(mx, p[i]);
    red[threadIdx.x] = mx; __syncthreads();
    for (int s = 128; s > 0; s >>= 1) {
        if (threadIdx.x < s) red[threadIdx.x] = fmaxf(red[threadIdx.x], red[threadIdx.x + s]);
        __syncthreads();
    }
    mx = red[0]; __syncthreads();

    float sum = 0.f;
    for (int i = threadIdx.x; i < V; i += 256) sum += __expf(p[i] - mx);
    red[threadIdx.x] = sum; __syncthreads();
    for (int s = 128; s > 0; s >>= 1) {
        if (threadIdx.x < s) red[threadIdx.x] += red[threadIdx.x + s];
        __syncthreads();
    }
    float lse = mx + __logf(red[0]);
    __syncthreads();

    for (int i = threadIdx.x; i < V; i += 256) p[i] = p[i] - lse;
}

// ---------------- host orchestration ----------------
extern "C" void kernel_launch(void* const* d_in, const int* in_sizes, int n_in,
                              void* d_out, int out_size, void* d_ws, size_t ws_size,
                              hipStream_t stream) {
    const int*   X       = (const int*)d_in[0];
    const int*   y       = (const int*)d_in[1];
    const float* emb_src = (const float*)d_in[2];
    const float* emb_tgt = (const float*)d_in[3];
    const float* enc_Wih = (const float*)d_in[4];
    const float* enc_Whh = (const float*)d_in[5];
    const float* enc_bih = (const float*)d_in[6];
    const float* enc_bhh = (const float*)d_in[7];
    const float* dec_Wih = (const float*)d_in[8];
    const float* dec_Whh = (const float*)d_in[9];
    const float* dec_bih = (const float*)d_in[10];
    const float* dec_bhh = (const float*)d_in[11];
    const float* fcW     = (const float*)d_in[12];
    const float* fcb     = (const float*)d_in[13];
    float* out = (float*)d_out;

    // ---- workspace carve-up ----
    char* ws = (char*)d_ws;
    size_t off = 0;
    auto take = [&](size_t bytes) -> void* {
        void* p = ws + off;
        off += (bytes + 255) & ~(size_t)255;
        return p;
    };
    const size_t WEL  = (size_t)2 * G4H * E_;        // elements per 2-layer LSTM weight tensor
    const size_t WSZ  = (size_t)G4H * E_;            // per-layer element stride
    const size_t FCEL = (size_t)VT_ * H_;

    __bf16* encWih_b = (__bf16*)take(WEL * sizeof(__bf16));
    __bf16* encWhh_b = (__bf16*)take(WEL * sizeof(__bf16));
    __bf16* decWih_b = (__bf16*)take(WEL * sizeof(__bf16));
    __bf16* decWhh_b = (__bf16*)take(WEL * sizeof(__bf16));
    __bf16* fcW_b    = (__bf16*)take(FCEL * sizeof(__bf16));
    // h: [layer][pingpong][B*H] bf16 ; c: [layer][B*H] f32 ; x: [B,E] bf16
    __bf16* h_buf    = (__bf16*)take((size_t)2 * 2 * B_ * H_ * sizeof(__bf16));
    float*  c_buf    = (float*)take((size_t)2 * B_ * H_ * sizeof(float));
    __bf16* x_buf    = (__bf16*)take((size_t)B_ * E_ * sizeof(__bf16));

    auto hb = [&](int l, int p) { return h_buf + ((size_t)(l * 2 + p)) * (B_ * H_); };
    auto cb = [&](int l)       { return c_buf + (size_t)l * (B_ * H_); };

    // ---- convert weights to bf16 (stay resident in 192MB L2 thereafter) ----
    f32_to_bf16_kernel<<<1024, 256, 0, stream>>>(enc_Wih, encWih_b, (int)WEL);
    f32_to_bf16_kernel<<<1024, 256, 0, stream>>>(enc_Whh, encWhh_b, (int)WEL);
    f32_to_bf16_kernel<<<1024, 256, 0, stream>>>(dec_Wih, decWih_b, (int)WEL);
    f32_to_bf16_kernel<<<1024, 256, 0, stream>>>(dec_Whh, decWhh_b, (int)WEL);
    f32_to_bf16_kernel<<<2048, 256, 0, stream>>>(fcW, fcW_b, (int)FCEL);

    // ---- zero initial h, c ----
    hipMemsetAsync(h_buf, 0, (size_t)2 * 2 * B_ * H_ * sizeof(__bf16), stream);
    hipMemsetAsync(c_buf, 0, (size_t)2 * B_ * H_ * sizeof(float), stream);

    const dim3 grid(H_ / 16, B_ / 16);   // 32 x 4
    const dim3 block(128);
    const int gthr = 256, gblk = (B_ * E_ + gthr - 1) / gthr;
    int pp = 0;

    // ---- encoder: 256 steps x 2 layers ----
    for (int t = 0; t < S_; ++t) {
        int np = pp ^ 1;
        gather_embed_kernel<<<gblk, gthr, 0, stream>>>(emb_src, X, S_, t, 0, x_buf);
        lstm_step_kernel<<<grid, block, 0, stream>>>(
            x_buf, encWih_b, encWhh_b, enc_bih, enc_bhh,
            hb(0, pp), cb(0), hb(0, np), cb(0));
        lstm_step_kernel<<<grid, block, 0, stream>>>(
            hb(0, np), encWih_b + WSZ, encWhh_b + WSZ, enc_bih + G4H, enc_bhh + G4H,
            hb(1, pp), cb(1), hb(1, np), cb(1));
        pp = np;
    }

    // ---- decoder: 32 teacher-forced steps x 2 layers ----
    for (int t = 0; t < T_; ++t) {
        int np = pp ^ 1;
        gather_embed_kernel<<<gblk, gthr, 0, stream>>>(emb_tgt, y, T_, t - 1,
                                                       (t == 0) ? 1 : 0, x_buf);
        lstm_step_kernel<<<grid, block, 0, stream>>>(
            x_buf, decWih_b, decWhh_b, dec_bih, dec_bhh,
            hb(0, pp), cb(0), hb(0, np), cb(0));
        lstm_step_kernel<<<grid, block, 0, stream>>>(
            hb(0, np), decWih_b + WSZ, decWhh_b + WSZ, dec_bih + G4H, dec_bhh + G4H,
            hb(1, pp), cb(1), hb(1, np), cb(1));
        pp = np;
    }

    // ---- final FC on last decoder top hidden state, then log-softmax ----
    fc_kernel<<<dim3(VT_ / 16, B_ / 16), 32, 0, stream>>>(hb(1, pp), fcW_b, fcb, out);
    log_softmax_kernel<<<B_, 256, 0, stream>>>(out, VT_);
}
